// SS2D_75058848465500
// MI455X (gfx1250) — compile-verified, matrix operand order checked by reading, not run on hardware
//
#include <hip/hip_runtime.h>
#include <stdint.h>

// ---------------- types ----------------
typedef __attribute__((ext_vector_type(16))) __bf16 v16bf;
typedef __attribute__((ext_vector_type(8)))  float  v8f;

union FragBF {
    v16bf v;
    uint32_t d[8];
};

// ---------------- helpers ----------------
__device__ __forceinline__ uint16_t f2bf(float f) {
    uint32_t u = __builtin_bit_cast(uint32_t, f);
    u += 0x7FFFu + ((u >> 16) & 1u);           // round-to-nearest-even
    return (uint16_t)(u >> 16);
}
__device__ __forceinline__ float bf2f(uint16_t h) {
    uint32_t u = ((uint32_t)h) << 16;
    return __builtin_bit_cast(float, u);
}
__device__ __forceinline__ float siluf(float v) {
    return v / (1.0f + __expf(-v));
}
__device__ __forceinline__ float softplusf(float v) {
    return (v > 20.0f) ? v : log1pf(__expf(v));
}

// problem constants
#define B_SZ   8
#define HH     96
#define WW     96
#define HW     9216           // 96*96
#define DMODEL 192
#define DIN    384            // d_inner
#define NSTATE 16
#define MTOK   73728          // 8*9216

// ---------------- weight conversion ----------------
__global__ __launch_bounds__(256) void cvt_f32_bf16(const float* __restrict__ s,
                                                    uint16_t* __restrict__ d, int n) {
    int i = blockIdx.x * 256 + threadIdx.x;
    if (i < n) d[i] = f2bf(s[i]);
}

// pack W_sB (384x16) and W_sC (384x16) into one (384x32) bf16 matrix [B | C]
__global__ __launch_bounds__(256) void pack_bc_w(const float* __restrict__ wb,
                                                 const float* __restrict__ wc,
                                                 uint16_t* __restrict__ d) {
    int i = blockIdx.x * 256 + threadIdx.x;
    if (i < DIN * NSTATE) {
        int k = i >> 4, n = i & 15;
        d[k * 32 + n]      = f2bf(wb[i]);
        d[k * 32 + 16 + n] = f2bf(wc[i]);
    }
}

// ---------------- WMMA GEMM: C[M,N] = A[M,K] @ W[K,N] (+epilogue) ----------------
// EPI: 0 = plain f32 store
//      1 = split: n<384 -> x_ssm f32 ; n>=384 -> silu -> bf16 gate
//      2 = softplus(v + bias) -> f32
//      3 = (v + bias) * gate_bf16 -> bf16
//      4 = transposed NCHW store: out[b, n, p] = v
// AXL : A is fp32 in x-layout (b, K, HW) (in-proj only); else A is bf16 row-major.
// WIDE: BN=64, each wave computes 16x32 (A-fragment reused over 2 B-fragments);
//       else BN=32, one 16x16 tile per wave.
template<int EPI, bool AXL, bool WIDE>
__global__ __launch_bounds__(256) void gemm_wmma(
    const float* __restrict__ Af, const uint16_t* __restrict__ Abf,
    const uint16_t* __restrict__ Wb,
    const float* __restrict__ bias, const uint16_t* __restrict__ gate,
    float* __restrict__ outF, uint16_t* __restrict__ outB,
    int M, int N, int K)
{
    constexpr int BM = 64;
    constexpr int BN = WIDE ? 64 : 32;
    constexpr int NF = WIDE ? 2 : 1;            // B fragments per wave
    constexpr int AP = (BM * 16) / 256;         // packed-dword A stores per thread (4)
    constexpr int WE = (BN * 32) / 256;         // W elements per thread (8 or 4)

    __shared__ __align__(16) uint16_t As[BM * 34];  // [m][k], row stride 34 (pad)
    __shared__ __align__(16) uint16_t Bs[BN * 34];  // [n][k], row stride 34 (pad)

    const int tid  = threadIdx.x;
    const int wv   = tid >> 5;
    const int lane = tid & 31;
    const int mt   = wv & 3;        // m-tile within block (0..3)
    const int nt   = wv >> 2;       // n-group within block (0..1)
    const int row0 = blockIdx.y * BM;
    const int n0   = blockIdx.x * BN;
    const int g    = lane >> 4;     // lane half selects K-group (ISA 16-bit A/B layout)
    const int mrow = (lane & 15) + mt * 16;

    // ---- hoisted per-thread staging coordinates (k0-invariant) ----
    const float*    aBaseF[AP];
    const uint16_t* aBaseH[AP];
    int aLds[AP];
    #pragma unroll
    for (int i = 0; i < AP; ++i) {
        int pe = tid + i * 256;
        if (AXL) {
            int m = pe & 63, k2 = (pe >> 6) << 1;    // m-fast: consecutive tid -> consecutive p
            int t = row0 + m;
            int bb = t / HW, p = t - bb * HW;
            aBaseF[i] = Af + (size_t)bb * (size_t)K * HW + (size_t)k2 * HW + p;
            aLds[i] = m * 34 + k2;
        } else {
            int m = pe >> 4, k2 = (pe & 15) << 1;    // k-fast: dword-contiguous global reads
            aBaseH[i] = Abf + (size_t)(row0 + m) * K + k2;
            aLds[i] = m * 34 + k2;
        }
    }
    const uint16_t* wBase[WE];
    int wLds[WE];
    #pragma unroll
    for (int i = 0; i < WE; ++i) {
        int e = tid + i * 256;
        int n = e & (BN - 1), kk = e / BN;
        wBase[i] = Wb + (size_t)kk * N + (n0 + n);
        wLds[i] = n * 34 + kk;
    }

    v8f acc[NF];
    #pragma unroll
    for (int f = 0; f < NF; ++f) acc[f] = (v8f){};

    const int nk = K >> 5;
    for (int ks = 0; ks < nk; ++ks) {
        const int k0 = ks << 5;

        // ---- gather phase: all global loads first (clause-friendly) ----
        float    av0[AP], av1[AP];
        uint32_t avp[AP];
        #pragma unroll
        for (int i = 0; i < AP; ++i) {
            if (AXL) {
                const float* ap = aBaseF[i] + (size_t)k0 * HW;
                av0[i] = ap[0];
                av1[i] = ap[HW];
            } else {
                avp[i] = *(const uint32_t*)(aBaseH[i] + k0);
            }
        }
        uint16_t wvv[WE];
        #pragma unroll
        for (int i = 0; i < WE; ++i) wvv[i] = wBase[i][(size_t)k0 * N];

        // ---- uniform prefetch of the next K slab ----
        if (ks + 1 < nk) {
            if (AXL) __builtin_prefetch(aBaseF[0] + (size_t)(k0 + 32) * HW, 0, 1);
            else     __builtin_prefetch(aBaseH[0] + k0 + 32, 0, 1);
            __builtin_prefetch(wBase[0] + (size_t)(k0 + 32) * N, 0, 1);
        }

        // ---- store phase: packed LDS writes ----
        #pragma unroll
        for (int i = 0; i < AP; ++i) {
            uint32_t pk = AXL ? ((uint32_t)f2bf(av0[i]) | ((uint32_t)f2bf(av1[i]) << 16))
                              : avp[i];
            *(uint32_t*)&As[aLds[i]] = pk;
        }
        #pragma unroll
        for (int i = 0; i < WE; ++i) Bs[wLds[i]] = wvv[i];
        __syncthreads();

        // Fragment fill per documented CDNA5 layout:
        //   element j: k = ((j&8)<<1) | (laneHalf<<3) | (j&7); read packed dword pairs.
        FragBF a;
        #pragma unroll
        for (int v = 0; v < 8; ++v) {
            int pidx = ((v & 4) << 1) | (g << 2) | (v & 3);   // dword index along K
            a.d[v] = *(const uint32_t*)&As[mrow * 34 + pidx * 2];
        }
        #pragma unroll
        for (int f = 0; f < NF; ++f) {
            const int ncol = (lane & 15) + nt * (16 * NF) + f * 16;
            FragBF b;
            #pragma unroll
            for (int v = 0; v < 8; ++v) {
                int pidx = ((v & 4) << 1) | (g << 2) | (v & 3);
                b.d[v] = *(const uint32_t*)&Bs[ncol * 34 + pidx * 2];
            }
            acc[f] = __builtin_amdgcn_wmma_f32_16x16x32_bf16(
                         false, a.v, false, b.v, (short)0, acc[f], false, false);
        }
        __syncthreads();
    }

    // D layout: lane n = L&15 ; VGPR r -> m = r + 8*(L>>4)
    #pragma unroll
    for (int f = 0; f < NF; ++f) {
        const int ncolg = n0 + nt * (16 * NF) + f * 16 + (lane & 15);
        #pragma unroll
        for (int r = 0; r < 8; ++r) {
            int m = mt * 16 + (g << 3) + r;
            int t = row0 + m;
            float v = acc[f][r];
            if (EPI == 0) {
                outF[(size_t)t * N + ncolg] = v;
            } else if (EPI == 1) {
                if (ncolg < DIN) outF[(size_t)t * DIN + ncolg] = v;
                else             outB[(size_t)t * DIN + (ncolg - DIN)] = f2bf(siluf(v));
            } else if (EPI == 2) {
                outF[(size_t)t * N + ncolg] = softplusf(v + bias[ncolg]);
            } else if (EPI == 3) {
                float s = (v + bias[ncolg]) * bf2f(gate[(size_t)t * DIN + ncolg]);
                outB[(size_t)t * N + ncolg] = f2bf(s);
            } else { // EPI == 4
                int bb = t / HW, p = t - bb * HW;
                outF[((size_t)bb * DMODEL + ncolg) * HW + p] = v;
            }
        }
    }
}

// ---------------- depthwise 3x3 conv + bias + SiLU (channels-last) -> bf16 ----------------
__global__ __launch_bounds__(256) void dwconv3x3_silu(const float* __restrict__ xin,
                                                      const float* __restrict__ cw,
                                                      const float* __restrict__ cb,
                                                      uint16_t* __restrict__ xout) {
    int idx = blockIdx.x * 256 + threadIdx.x;
    if (idx >= MTOK * DIN) return;
    int c = idx % DIN;
    int t = idx / DIN;
    int b = t / HW;
    int p = t - b * HW;
    int h = p / WW, w = p - h * WW;
    float acc = cb[c];
    #pragma unroll
    for (int dh = -1; dh <= 1; ++dh) {
        int hh = h + dh;
        if (hh < 0 || hh >= HH) continue;
        #pragma unroll
        for (int dw = -1; dw <= 1; ++dw) {
            int ww2 = w + dw;
            if (ww2 < 0 || ww2 >= WW) continue;
            acc += xin[((size_t)b * HW + hh * WW + ww2) * DIN + c] *
                   cw[c * 9 + (dh + 1) * 3 + (dw + 1)];
        }
    }
    xout[idx] = f2bf(siluf(acc));
}

// ---------------- 4-direction selective scan ----------------
// grid (768, 4): blockIdx.x = scan line (b*96 + h or b*96 + w), blockIdx.y = direction
// dir 0: row fwd, 1: row bwd, 2: col fwd, 3: col bwd (matches concat order y1..y4)
// The whole line's [B|C] (96 x 32 f32 = 12 KB) is staged in LDS once -> the 96
// sequential steps then run barrier-free with LDS broadcast reads.
__global__ __launch_bounds__(384) void scan4(const uint16_t* __restrict__ xc,
                                             const float* __restrict__ delta,
                                             const float* __restrict__ bc,
                                             const float* __restrict__ A_log,
                                             const float* __restrict__ Dp,
                                             uint16_t* __restrict__ ym) {
    const int d   = threadIdx.x;          // channel 0..383
    const int dir = blockIdx.y;
    const int rb  = blockIdx.x;
    const int b   = rb / 96;
    const int q   = rb - b * 96;          // h for row scans, w for col scans

    const int tbase = b * HW + ((dir < 2) ? q * WW : q);
    const int tstep = (dir < 2) ? 1 : WW;

    __shared__ float sBC[96 * 32];
    for (int i = d; i < 96 * 32; i += 384) {
        int pos = i >> 5, j = i & 31;
        sBC[i] = bc[(size_t)(tbase + pos * tstep) * 32 + j];
    }
    __syncthreads();

    float An[NSTATE];
    #pragma unroll
    for (int n = 0; n < NSTATE; ++n) An[n] = -__expf(A_log[n]);
    const float Dd = Dp[d];

    float hst[NSTATE];
    #pragma unroll
    for (int n = 0; n < NSTATE; ++n) hst[n] = 0.0f;

    for (int s = 0; s < 96; ++s) {
        const int pos = (dir & 1) ? (95 - s) : s;
        const int t = tbase + pos * tstep;
        const float* bcp = &sBC[pos * 32];

        const float xd = bf2f(xc[(size_t)t * DIN + d]);
        const float dt = delta[(size_t)t * DIN + d];
        float y = 0.0f;
        #pragma unroll
        for (int n = 0; n < NSTATE; ++n) {
            float ab = __expf(dt * An[n]);
            hst[n] = ab * hst[n] + (dt * bcp[n]) * xd;
            y += hst[n] * bcp[16 + n];
        }
        y += xd * Dd;
        ym[(size_t)t * (4 * DIN) + dir * DIN + d] = f2bf(y);
    }
}

// ---------------- host orchestration ----------------
extern "C" void kernel_launch(void* const* d_in, const int* in_sizes, int n_in,
                              void* d_out, int out_size, void* d_ws, size_t ws_size,
                              hipStream_t stream) {
    const float* x       = (const float*)d_in[0];
    const float* W_in    = (const float*)d_in[1];
    const float* conv_w  = (const float*)d_in[2];
    const float* conv_b  = (const float*)d_in[3];
    const float* W_sB    = (const float*)d_in[4];
    const float* W_sC    = (const float*)d_in[5];
    const float* W_sd    = (const float*)d_in[6];
    const float* b_sd    = (const float*)d_in[7];
    const float* W_merge = (const float*)d_in[8];
    const float* b_merge = (const float*)d_in[9];
    const float* W_out   = (const float*)d_in[10];
    const float* A_log   = (const float*)d_in[11];
    const float* Dp      = (const float*)d_in[12];
    float* out = (float*)d_out;

    char* wsp = (char*)d_ws;
    auto alloc = [&](size_t bytes) -> char* {
        char* p = wsp;
        wsp += (bytes + 255) & ~(size_t)255;
        return p;
    };
    const size_t M = MTOK;
    float*    x_ssm   = (float*)alloc(M * DIN * 4);       // in-proj ssm half, f32
    uint16_t* gate_bf = (uint16_t*)alloc(M * DIN * 2);    // silu(gate), bf16
    uint16_t* xc_bf   = (uint16_t*)alloc(M * DIN * 2);    // conv+silu, bf16
    float*    delta   = (float*)alloc(M * DIN * 4);       // softplus delta, f32
    float*    bcbuf   = (float*)alloc(M * 32 * 4);        // [B|C] per token, f32
    uint16_t* ym      = (uint16_t*)alloc(M * 4 * DIN * 2);// 4-dir scan outputs, bf16
    uint16_t* yg      = (uint16_t*)alloc(M * DIN * 2);    // gated merge out, bf16
    uint16_t* w_in_bf  = (uint16_t*)alloc((size_t)DMODEL * 2 * DIN * 2);
    uint16_t* w_bc_bf  = (uint16_t*)alloc((size_t)DIN * 32 * 2);
    uint16_t* w_sd_bf  = (uint16_t*)alloc((size_t)DIN * DIN * 2);
    uint16_t* w_mg_bf  = (uint16_t*)alloc((size_t)4 * DIN * DIN * 2);
    uint16_t* w_out_bf = (uint16_t*)alloc((size_t)DIN * DMODEL * 2);

    // --- weights -> bf16 ---
    int n1 = DMODEL * 2 * DIN;
    cvt_f32_bf16<<<(n1 + 255) / 256, 256, 0, stream>>>(W_in, w_in_bf, n1);
    int n2 = DIN * DIN;
    cvt_f32_bf16<<<(n2 + 255) / 256, 256, 0, stream>>>(W_sd, w_sd_bf, n2);
    int n3 = 4 * DIN * DIN;
    cvt_f32_bf16<<<(n3 + 255) / 256, 256, 0, stream>>>(W_merge, w_mg_bf, n3);
    int n4 = DIN * DMODEL;
    cvt_f32_bf16<<<(n4 + 255) / 256, 256, 0, stream>>>(W_out, w_out_bf, n4);
    pack_bc_w<<<(DIN * NSTATE + 255) / 256, 256, 0, stream>>>(W_sB, W_sC, w_bc_bf);

    const dim3 blk(256);
    const int MB = (int)(M / 64);

    // 1) in-proj: x(b,192,hw) @ W_in(192,768) -> x_ssm f32 | silu(gate) bf16
    gemm_wmma<1, true, true><<<dim3(2 * DIN / 64, MB), blk, 0, stream>>>(
        x, nullptr, w_in_bf, nullptr, nullptr, x_ssm, gate_bf, (int)M, 2 * DIN, DMODEL);

    // 2) depthwise 3x3 conv + SiLU -> bf16 channels-last
    dwconv3x3_silu<<<((int)(M * DIN) + 255) / 256, 256, 0, stream>>>(
        x_ssm, conv_w, conv_b, xc_bf);

    // 3) delta = softplus(xc @ W_sd + b_sd)   (computed once; row/col sel() coincide)
    gemm_wmma<2, false, true><<<dim3(DIN / 64, MB), blk, 0, stream>>>(
        nullptr, xc_bf, w_sd_bf, b_sd, nullptr, delta, nullptr, (int)M, DIN, DIN);

    // 4) [B|C] = xc @ [W_sB|W_sC]  (N=32 -> narrow tile)
    gemm_wmma<0, false, false><<<dim3(1, MB), blk, 0, stream>>>(
        nullptr, xc_bf, w_bc_bf, nullptr, nullptr, bcbuf, nullptr, (int)M, 32, DIN);

    // 5) 4-direction selective scan -> ym (b,hw,1536) bf16
    scan4<<<dim3(B_SZ * 96, 4), dim3(DIN), 0, stream>>>(
        xc_bf, delta, bcbuf, A_log, Dp, ym);

    // 6) merge: (ym @ W_merge + b_merge) * silu(gate) -> yg bf16
    gemm_wmma<3, false, true><<<dim3(DIN / 64, MB), blk, 0, stream>>>(
        nullptr, ym, w_mg_bf, b_merge, gate_bf, nullptr, yg, (int)M, DIN, 4 * DIN);

    // 7) out-proj with transposed NCHW store -> d_out f32 (b,192,96,96)
    gemm_wmma<4, false, true><<<dim3(DMODEL / 64, MB), blk, 0, stream>>>(
        nullptr, yg, w_out_bf, nullptr, nullptr, out, nullptr, (int)M, DMODEL, DIN);

    (void)in_sizes; (void)n_in; (void)out_size; (void)ws_size;
}
